// TransformerLayer_51281909514575
// MI455X (gfx1250) — compile-verified, matrix operand order checked by reading, not run on hardware
//
#include <hip/hip_runtime.h>
#include <math.h>

// ---------------------------------------------------------------------------
// Types for CDNA5 WMMA (wave32): v_wmma_f32_16x16x32_bf16
// ---------------------------------------------------------------------------
typedef __bf16 bf16;
typedef bf16  v16bf __attribute__((ext_vector_type(16)));
typedef bf16  v8bf  __attribute__((ext_vector_type(8)));
typedef float v8f   __attribute__((ext_vector_type(8)));

#define WMMA_BF16(A, B, C) \
    __builtin_amdgcn_wmma_f32_16x16x32_bf16(false, (A), false, (B), (short)0, (C), false, false)

// Build a 16x32 A/B fragment for one lane: elements 0..7 from `lo`,
// 8..15 from `hi` (K and K+16 halves per the ISA 16-bit A-matrix layout).
static __device__ __forceinline__ v16bf frag16(const bf16* lo, const bf16* hi) {
    v8bf a = *(const v8bf*)lo;
    v8bf b = *(const v8bf*)hi;
    v16bf f;
#pragma unroll
    for (int i = 0; i < 8; ++i) { f[i] = a[i]; f[i + 8] = b[i]; }
    return f;
}

enum { OUT_BF16 = 0, OUT_TRV = 1, OUT_F32 = 2 };
enum { EPI_NONE = 0, EPI_GELU = 1 };

// ---------------------------------------------------------------------------
// Generic WMMA GEMM:  Out = epi(A[M,K] * W[K,Ntot] + bias)
// Block = 256 threads (8 waves), tile 128(M) x 128(N), K-step 32,
// double-buffered LDS staging (one barrier per k-step, global loads for the
// next k-block overlap the current block's 8 WMMAs per wave).
// Wave grid 4(M) x 2(N); each wave computes a 32x64 patch (2x4 WMMA tiles).
// ---------------------------------------------------------------------------
template<int K, bool A_BF16, bool HAS_BIAS, int EPI, int OUTMODE>
__global__ __launch_bounds__(256)
void gemm_bf16_wmma(const void* __restrict__ Aptr, const float* __restrict__ W,
                    const float* __restrict__ bias, void* __restrict__ Out, int Ntot)
{
    __shared__ __align__(16) bf16 Asl[2][128][40];  // padded: 16B-aligned, bank-spread
    __shared__ __align__(16) bf16 Wsl[2][128][40];  // W tile transposed [n][k]

    const int t    = threadIdx.x;
    const int lane = t & 31;
    const int w    = t >> 5;
    const int wm   = (w & 3) * 32;
    const int wn   = (w >> 2) * 64;
    const int lrow = lane & 15;
    const int lhi  = lane >> 4;
    const int kb   = lhi * 8;
    const int m0   = blockIdx.x * 128;
    const int n0   = blockIdx.y * 128;

    // stage one 128x32 A block + 32x128 W block into LDS buffer `buf`
    auto stage = [&](int buf, int k0) {
        {   // A tile: thread -> 16 contiguous elements of one row
            const int m  = t >> 1;
            const int kh = (t & 1) * 16;
            if (A_BF16) {
                const bf16* src = (const bf16*)Aptr + (size_t)(m0 + m) * K + k0 + kh;
                *(v8bf*)&Asl[buf][m][kh]     = *(const v8bf*)(src);
                *(v8bf*)&Asl[buf][m][kh + 8] = *(const v8bf*)(src + 8);
            } else {
                const float* src = (const float*)Aptr + (size_t)(m0 + m) * K + k0 + kh;
                v8bf o0, o1;
#pragma unroll
                for (int i = 0; i < 8; ++i) { o0[i] = (bf16)src[i]; o1[i] = (bf16)src[i + 8]; }
                *(v8bf*)&Asl[buf][m][kh]     = o0;
                *(v8bf*)&Asl[buf][m][kh + 8] = o1;
            }
        }
        {   // W tile 32x128 (coalesced along n), stored transposed [n][k]
            const int kk = t >> 3;
            const int nn = (t & 7) * 16;
            const float* src = W + (size_t)(k0 + kk) * Ntot + n0 + nn;
#pragma unroll
            for (int i = 0; i < 16; ++i) Wsl[buf][nn + i][kk] = (bf16)src[i];
        }
    };

    v8f acc[2][4];
#pragma unroll
    for (int i = 0; i < 2; ++i)
#pragma unroll
        for (int j = 0; j < 4; ++j)
#pragma unroll
            for (int r = 0; r < 8; ++r) acc[i][j][r] = 0.0f;

    stage(0, 0);
    __syncthreads();

    for (int k0 = 0; k0 < K; k0 += 32) {
        const int cur = (k0 >> 5) & 1;
        if (k0 + 32 < K) stage(cur ^ 1, k0 + 32);   // overlaps WMMAs below

        v16bf af[2], bfr[4];
#pragma unroll
        for (int i = 0; i < 2; ++i) {
            const int r = wm + i * 16 + lrow;
            af[i] = frag16(&Asl[cur][r][kb], &Asl[cur][r][kb + 16]);
        }
#pragma unroll
        for (int j = 0; j < 4; ++j) {
            const int c = wn + j * 16 + lrow;
            bfr[j] = frag16(&Wsl[cur][c][kb], &Wsl[cur][c][kb + 16]);
        }
#pragma unroll
        for (int i = 0; i < 2; ++i)
#pragma unroll
            for (int j = 0; j < 4; ++j)
                acc[i][j] = WMMA_BF16(af[i], bfr[j], acc[i][j]);
        __syncthreads();
    }

    // Epilogue. C/D layout: n = lane&15, m = vgpr + 8*(lane>>4).
#pragma unroll
    for (int i = 0; i < 2; ++i) {
#pragma unroll
        for (int j = 0; j < 4; ++j) {
            const int n = n0 + wn + j * 16 + lrow;
            float bvv = 0.0f;
            if (HAS_BIAS) bvv = bias[n];
#pragma unroll
            for (int r = 0; r < 8; ++r) {
                const int m = m0 + wm + i * 16 + r + lhi * 8;
                float v = acc[i][j][r] + bvv;
                if (EPI == EPI_GELU) v = 0.5f * v * (1.0f + erff(v * 0.70710678118654752f));
                if (OUTMODE == OUT_BF16) {
                    ((bf16*)Out)[(size_t)m * Ntot + n] = (bf16)v;
                } else if (OUTMODE == OUT_TRV) {   // V written transposed: [B][D=256][S=4096]
                    const int bidx = m >> 12;
                    const int s    = m & 4095;
                    ((bf16*)Out)[((size_t)(bidx * 256 + n)) * 4096 + s] = (bf16)v;
                } else {
                    ((float*)Out)[(size_t)m * Ntot + n] = v;
                }
            }
        }
    }
}

// ---------------------------------------------------------------------------
// Flash attention: 8 waves/block, 16 query rows per wave, 32-key S-blocks.
// Q,K: [B*4096, 256] bf16 row-major; Vt: [B][256][4096] bf16; O: [B*4096,256] bf16.
// QK: two independent accumulator chains (score tiles) sharing each Q frag,
// K-frags double-buffered. PV: 16 independent chains, V-frags 4-deep pipelined.
// ---------------------------------------------------------------------------
__global__ __launch_bounds__(256)
void flash_attn(const bf16* __restrict__ Q, const bf16* __restrict__ Kmat,
                const bf16* __restrict__ Vt, bf16* __restrict__ O)
{
    __shared__ __align__(16) float Sbuf[8][16][33];
    __shared__ __align__(16) bf16  Pbuf[8][16][40];
    __shared__ float Cbuf[8][16];

    const int t    = threadIdx.x;
    const int lane = t & 31;
    const int w    = t >> 5;
    const int lrow = lane & 15;
    const int lhi  = lane >> 4;
    const int kb   = lhi * 8;
    const int qrow = blockIdx.x * 128 + w * 16;   // global query row
    const int bidx = qrow >> 12;                  // L = 4096
    const float scale = 0.0625f;                  // 1/sqrt(256)

    // Cache all Q fragments for these 16 rows (8 K-steps over D=256)
    v16bf qf[8];
    {
        const bf16* qp = Q + (size_t)(qrow + lrow) * 256;
#pragma unroll
        for (int kk = 0; kk < 8; ++kk)
            qf[kk] = frag16(qp + kk * 32 + kb, qp + kk * 32 + kb + 16);
    }

    v8f o[16];   // 16x256 f32 output accumulators
#pragma unroll
    for (int i = 0; i < 16; ++i)
#pragma unroll
        for (int r = 0; r < 8; ++r) o[i][r] = 0.0f;

    float m_run = -1e30f;   // valid in lanes 0..15 (one query row per lane)
    float l_run = 0.0f;

    const bf16* Kb = Kmat + (size_t)bidx * 4096 * 256;
    const bf16* Vb = Vt   + (size_t)bidx * 256 * 4096;

    for (int s0 = 0; s0 < 4096; s0 += 32) {
        if (s0 + 32 < 4096)   // global_prefetch of next K block
            __builtin_prefetch(Kb + (size_t)(s0 + 32 + lrow) * 256 + kb, 0, 1);

        // --- scores: 16 queries x 32 keys; two interleaved WMMA chains ---
        {
            const bf16* kp0 = Kb + (size_t)(s0 + lrow) * 256;
            const bf16* kp1 = kp0 + 16 * 256;
            v8f sc0, sc1;
#pragma unroll
            for (int r = 0; r < 8; ++r) { sc0[r] = 0.0f; sc1[r] = 0.0f; }
            v16bf kf0[2], kf1[2];
#pragma unroll
            for (int i = 0; i < 2; ++i) {
                kf0[i] = frag16(kp0 + i * 32 + kb, kp0 + i * 32 + kb + 16);
                kf1[i] = frag16(kp1 + i * 32 + kb, kp1 + i * 32 + kb + 16);
            }
#pragma unroll
            for (int kk = 0; kk < 8; ++kk) {
                sc0 = WMMA_BF16(qf[kk], kf0[kk & 1], sc0);
                sc1 = WMMA_BF16(qf[kk], kf1[kk & 1], sc1);
                if (kk + 2 < 8) {   // refill 2 ahead
                    kf0[kk & 1] = frag16(kp0 + (kk + 2) * 32 + kb, kp0 + (kk + 2) * 32 + kb + 16);
                    kf1[kk & 1] = frag16(kp1 + (kk + 2) * 32 + kb, kp1 + (kk + 2) * 32 + kb + 16);
                }
            }
#pragma unroll
            for (int r = 0; r < 8; ++r) {
                Sbuf[w][r + lhi * 8][lrow]      = sc0[r] * scale;
                Sbuf[w][r + lhi * 8][16 + lrow] = sc1[r] * scale;
            }
        }
        __syncthreads();

        // --- online softmax: lane i (0..15) owns query row i of this wave ---
        if (lane < 16) {
            float sv[32];
            float bmax = -1e30f;
#pragma unroll
            for (int j = 0; j < 32; ++j) { sv[j] = Sbuf[w][lane][j]; bmax = fmaxf(bmax, sv[j]); }
            const float mn = fmaxf(m_run, bmax);
            const float c  = __expf(m_run - mn);
            float ssum = 0.0f;
#pragma unroll
            for (int j = 0; j < 32; ++j) {
                const float p = __expf(sv[j] - mn);
                ssum += p;
                Pbuf[w][lane][j] = (bf16)p;
            }
            l_run = c * l_run + ssum;
            m_run = mn;
            Cbuf[w][lane] = c;
        }
        __syncthreads();

        // --- rescale O by correction factors, then O += P * V ---
        float crow[8];
#pragma unroll
        for (int r = 0; r < 8; ++r) crow[r] = Cbuf[w][r + lhi * 8];
#pragma unroll
        for (int i = 0; i < 16; ++i)
#pragma unroll
            for (int r = 0; r < 8; ++r) o[i][r] *= crow[r];

        const v16bf pf = frag16(&Pbuf[w][lrow][kb], &Pbuf[w][lrow][kb + 16]);
        v16bf vf[4];   // 4-deep V-fragment pipeline
#pragma unroll
        for (int i = 0; i < 4; ++i) {
            const bf16* vp = Vb + (size_t)(i * 16 + lrow) * 4096 + s0;
            vf[i] = frag16(vp + kb, vp + kb + 16);
        }
#pragma unroll
        for (int nt = 0; nt < 16; ++nt) {
            o[nt] = WMMA_BF16(pf, vf[nt & 3], o[nt]);
            if (nt + 4 < 16) {
                const bf16* vp = Vb + (size_t)((nt + 4) * 16 + lrow) * 4096 + s0;
                vf[nt & 3] = frag16(vp + kb, vp + kb + 16);
            }
        }
        __syncthreads();
    }

    // --- final 1/l normalization and store ---
    if (lane < 16) Cbuf[w][lane] = 1.0f / l_run;
    __syncthreads();
    float crow[8];
#pragma unroll
    for (int r = 0; r < 8; ++r) crow[r] = Cbuf[w][r + lhi * 8];
#pragma unroll
    for (int nt = 0; nt < 16; ++nt) {
#pragma unroll
        for (int r = 0; r < 8; ++r) {
            const int m = qrow + r + lhi * 8;
            const int d = nt * 16 + lrow;
            O[(size_t)m * 256 + d] = (bf16)(o[nt][r] * crow[r]);
        }
    }
}

// ---------------------------------------------------------------------------
// LayerNorm over D=256: one 256-thread block per row.
// ---------------------------------------------------------------------------
__global__ __launch_bounds__(256)
void ln_store(const float* __restrict__ X, const float* __restrict__ g,
              const float* __restrict__ beta, bf16* __restrict__ bigA)
{
    __shared__ float red[256];
    const int row = blockIdx.x;
    const int t   = threadIdx.x;
    const float x = X[(size_t)row * 256 + t];
    red[t] = x;
    __syncthreads();
#pragma unroll
    for (int s = 128; s > 0; s >>= 1) { if (t < s) red[t] += red[t + s]; __syncthreads(); }
    const float mu = red[0] * (1.0f / 256.0f);
    __syncthreads();
    const float d = x - mu;
    red[t] = d * d;
    __syncthreads();
#pragma unroll
    for (int s = 128; s > 0; s >>= 1) { if (t < s) red[t] += red[t + s]; __syncthreads(); }
    const float var = red[0] * (1.0f / 256.0f);
    const float y = d * rsqrtf(var + 1e-5f) * g[t] + beta[t];
    bigA[(size_t)row * 512 + 256 + t] = (bf16)y;   // right half of the concat buffer
}

__global__ __launch_bounds__(256)
void ln_residual(const float* __restrict__ X, const float* __restrict__ g,
                 const float* __restrict__ beta, const float* __restrict__ src,
                 float* __restrict__ out)
{
    __shared__ float red[256];
    const int row = blockIdx.x;
    const int t   = threadIdx.x;
    const float x = X[(size_t)row * 256 + t];
    red[t] = x;
    __syncthreads();
#pragma unroll
    for (int s = 128; s > 0; s >>= 1) { if (t < s) red[t] += red[t + s]; __syncthreads(); }
    const float mu = red[0] * (1.0f / 256.0f);
    __syncthreads();
    const float d = x - mu;
    red[t] = d * d;
    __syncthreads();
#pragma unroll
    for (int s = 128; s > 0; s >>= 1) { if (t < s) red[t] += red[t + s]; __syncthreads(); }
    const float var = red[0] * (1.0f / 256.0f);
    const float y = d * rsqrtf(var + 1e-5f) * g[t] + beta[t];
    out[(size_t)row * 256 + t] = src[(size_t)row * 256 + t] + y;
}

__global__ __launch_bounds__(256)
void src_to_bigA(const float* __restrict__ src, bf16* __restrict__ bigA)
{
    const int row = blockIdx.x;
    const int t   = threadIdx.x;
    bigA[(size_t)row * 512 + t] = (bf16)src[(size_t)row * 256 + t];  // left half of concat
}

// ---------------------------------------------------------------------------
// Orchestration. Workspace (8 MiB units, overlaid):
//  [0]: Qbf   -> msg_raw (f32, 2 units) -> yraw (f32, 2 units)
//  [1]: Kbf
//  [2]: Vt    -> bigA (2 units)
//  [3]: Obf
//  [4..5]: h  (bf16 [M,512])
// Peak: 48 MiB.
// ---------------------------------------------------------------------------
extern "C" void kernel_launch(void* const* d_in, const int* in_sizes, int n_in,
                              void* d_out, int out_size, void* d_ws, size_t ws_size,
                              hipStream_t stream)
{
    const float* source = (const float*)d_in[0];
    const float* target = (const float*)d_in[1];
    const float* Wq = (const float*)d_in[2];
    const float* bq = (const float*)d_in[3];
    const float* Wk = (const float*)d_in[4];
    const float* bk = (const float*)d_in[5];
    const float* Wv = (const float*)d_in[6];
    const float* bv = (const float*)d_in[7];
    const float* Wm = (const float*)d_in[8];
    const float* bm = (const float*)d_in[9];
    const float* g1 = (const float*)d_in[10];
    const float* beta1 = (const float*)d_in[11];
    const float* W1 = (const float*)d_in[12];
    const float* W2 = (const float*)d_in[13];
    const float* g2 = (const float*)d_in[14];
    const float* beta2 = (const float*)d_in[15];
    float* out = (float*)d_out;

    char* base = (char*)d_ws;
    const size_t SZ = (size_t)16384 * 256 * 2;      // 8 MiB
    bf16*  Qbf     = (bf16*)(base + 0 * SZ);
    bf16*  Kbf     = (bf16*)(base + 1 * SZ);
    bf16*  Vtb     = (bf16*)(base + 2 * SZ);
    bf16*  Obf     = (bf16*)(base + 3 * SZ);
    float* msg_raw = (float*)(base + 0 * SZ);       // over Q,K (dead after attention)
    bf16*  bigA    = (bf16*)(base + 2 * SZ);        // over Vt,Obf (dead after msg GEMM)
    bf16*  hbuf    = (bf16*)(base + 4 * SZ);
    float* yraw    = (float*)(base + 0 * SZ);       // over msg_raw (dead after LN1)

    const dim3 blk(256);

    // Q/K/V projections (f32 in, bf16 out; V transposed for PV fragment loads)
    gemm_bf16_wmma<256, false, true, EPI_NONE, OUT_BF16><<<dim3(128, 2), blk, 0, stream>>>(source, Wq, bq, Qbf, 256);
    gemm_bf16_wmma<256, false, true, EPI_NONE, OUT_BF16><<<dim3(128, 2), blk, 0, stream>>>(target, Wk, bk, Kbf, 256);
    gemm_bf16_wmma<256, false, true, EPI_NONE, OUT_TRV ><<<dim3(128, 2), blk, 0, stream>>>(target, Wv, bv, Vtb, 256);

    // Fused softmax(QK^T)V
    flash_attn<<<dim3(128), blk, 0, stream>>>(Qbf, Kbf, Vtb, Obf);

    // message @ Wm + bm (raw f32, LN needs full rows)
    gemm_bf16_wmma<256, true, true, EPI_NONE, OUT_F32><<<dim3(128, 2), blk, 0, stream>>>(Obf, Wm, bm, msg_raw, 256);

    // concat(source, LN1(message)) staged as bf16 [M,512]
    src_to_bigA<<<dim3(16384), blk, 0, stream>>>(source, bigA);
    ln_store<<<dim3(16384), blk, 0, stream>>>(msg_raw, g1, beta1, bigA);

    // MLP: gelu(h@W1) then @W2
    gemm_bf16_wmma<512, true, false, EPI_GELU, OUT_BF16><<<dim3(128, 4), blk, 0, stream>>>(bigA, W1, nullptr, hbuf, 512);
    gemm_bf16_wmma<512, true, false, EPI_NONE, OUT_F32 ><<<dim3(128, 2), blk, 0, stream>>>(hbuf, W2, nullptr, yraw, 256);

    // LN2 + residual -> d_out
    ln_residual<<<dim3(16384), blk, 0, stream>>>(yraw, g2, beta2, source, out);

    (void)in_sizes; (void)n_in; (void)out_size; (void)ws_size;
}
